// MultiheadLatentAttention_50414326121217
// MI455X (gfx1250) — matrix operand order repaired and checked by reading, not compiled, not run on hardware
//
#include <hip/hip_runtime.h>
#include <hip/hip_bf16.h>

#define B     4
#define LSEQ  4096
#define DIMQ  7168
#define H     128
#define KDIM  128
#define DC    512
#define DCQ   1536
#define RR    64

typedef __attribute__((ext_vector_type(16))) __bf16 v16bf;
typedef __attribute__((ext_vector_type(8)))  __bf16 v8bf;
typedef __attribute__((ext_vector_type(8)))  float  v8f;
typedef __attribute__((ext_vector_type(4)))  float  f4;

// Non-temporal 16B load for single-use weight streams (keeps the L2-resident
// attention working set hot; lowers to global_load_b128 th:TH_LOAD_NT).
__device__ __forceinline__ f4 ntload4(const float* p) {
    return __builtin_nontemporal_load((const f4*)p);
}
__device__ __forceinline__ f4 ld4(const float* p) {
    return *(const f4*)p;
}

// ---------------------------------------------------------------------------
// 1) RoPE on kv_c, emit bf16 in [B,L,DC] (row-major over d) and [B,DC,L]
//    (row-major over l) layouts for the two WMMA consumers.
// ---------------------------------------------------------------------------
__global__ __launch_bounds__(512)
void rope_kernel(const float* __restrict__ kv_c,
                 __bf16* __restrict__ kv_bf,
                 __bf16* __restrict__ kvT_bf) {
    const int l = blockIdx.x;
    const int b = blockIdx.y;
    const int d = threadIdx.x;                       // 0..511
    const float* row = kv_c + ((size_t)b * LSEQ + l) * DC;
    float x   = row[d];
    float rot = (d < DC / 2) ? -row[d + DC / 2] : row[d - DC / 2];
    // inv_freq[i] = 10000^(-2i/DC), i = d>>1 ; sin/cos repeated pairwise
    float expo = (float)(2 * (d >> 1)) / (float)DC;
    float invf = __expf(-__logf(10000.0f) * expo);
    float ang  = (float)l * invf;
    float v    = x * __cosf(ang) + rot * __sinf(ang);
    __bf16 bv = (__bf16)v;
    kv_bf [((size_t)b * LSEQ + l) * DC + d] = bv;
    kvT_bf[((size_t)b * DC   + d) * LSEQ + l] = bv;
}

// ---------------------------------------------------------------------------
// 2) q_hk[b,n] = hidden_q[b,:] . Wq[n,:]   (wave per output column n)
//    Wq is a 470 MB single-use stream -> non-temporal loads.
// ---------------------------------------------------------------------------
__global__ __launch_bounds__(256)
void qhk_kernel(const float* __restrict__ hidden, const float* __restrict__ Wq,
                float* __restrict__ q_hk) {
    const int wv = threadIdx.x >> 5, lane = threadIdx.x & 31;
    const int n = blockIdx.x * 8 + wv;               // 0..H*KDIM-1
    const float* wrow = Wq + (size_t)n * DIMQ;
    float a0 = 0.f, a1 = 0.f, a2 = 0.f, a3 = 0.f;
    for (int d = lane * 4; d < DIMQ; d += 128) {
        f4 w  = ntload4(wrow + d);
        f4 h0 = ld4(hidden + 0 * DIMQ + d);
        f4 h1 = ld4(hidden + 1 * DIMQ + d);
        f4 h2 = ld4(hidden + 2 * DIMQ + d);
        f4 h3 = ld4(hidden + 3 * DIMQ + d);
        a0 += w.x * h0.x + w.y * h0.y + w.z * h0.z + w.w * h0.w;
        a1 += w.x * h1.x + w.y * h1.y + w.z * h1.z + w.w * h1.w;
        a2 += w.x * h2.x + w.y * h2.y + w.z * h2.z + w.w * h2.w;
        a3 += w.x * h3.x + w.y * h3.y + w.z * h3.z + w.w * h3.w;
    }
    #pragma unroll
    for (int off = 16; off > 0; off >>= 1) {
        a0 += __shfl_down(a0, off);
        a1 += __shfl_down(a1, off);
        a2 += __shfl_down(a2, off);
        a3 += __shfl_down(a3, off);
    }
    if (lane == 0) {
        q_hk[0 * (H * KDIM) + n] = a0;
        q_hk[1 * (H * KDIM) + n] = a1;
        q_hk[2 * (H * KDIM) + n] = a2;
        q_hk[3 * (H * KDIM) + n] = a3;
    }
}

// ---------------------------------------------------------------------------
// 3) q_big[b,h,q] = sum_k q_hk[b,h,k] * w_kc_q[h,k,q]   (100 MB NT stream)
// ---------------------------------------------------------------------------
__global__ __launch_bounds__(256)
void qbig_kernel(const float* __restrict__ q_hk, const float* __restrict__ w_kc_q,
                 float* __restrict__ q_big) {
    __shared__ float qs[B][KDIM];
    const int h = blockIdx.y;
    const int q = blockIdx.x * 256 + threadIdx.x;    // 0..DCQ-1
    for (int i = threadIdx.x; i < B * KDIM; i += 256)
        qs[i >> 7][i & 127] = q_hk[(size_t)(i >> 7) * (H * KDIM) + h * KDIM + (i & 127)];
    __syncthreads();
    float a0 = 0.f, a1 = 0.f, a2 = 0.f, a3 = 0.f;
    const float* wp = w_kc_q + ((size_t)h * KDIM) * DCQ + q;
    for (int k = 0; k < KDIM; ++k) {
        float w = __builtin_nontemporal_load(wp + (size_t)k * DCQ);
        a0 += qs[0][k] * w; a1 += qs[1][k] * w;
        a2 += qs[2][k] * w; a3 += qs[3][k] * w;
    }
    q_big[((size_t)0 * H + h) * DCQ + q] = a0;
    q_big[((size_t)1 * H + h) * DCQ + q] = a1;
    q_big[((size_t)2 * H + h) * DCQ + q] = a2;
    q_big[((size_t)3 * H + h) * DCQ + q] = a3;
}

// ---------------------------------------------------------------------------
// 4) q_r[b,h,r] = sum_q q_big[b,h,q] * W_qr[h,q,r]   (50 MB NT stream)
// ---------------------------------------------------------------------------
__global__ __launch_bounds__(64)
void qr_kernel(const float* __restrict__ q_big, const float* __restrict__ W_qr,
               float* __restrict__ q_r) {
    const int h = blockIdx.x;
    const int r = threadIdx.x;                       // 0..63
    const float* wp = W_qr + (size_t)h * DCQ * RR + r;
    const float* b0 = q_big + ((size_t)0 * H + h) * DCQ;
    const float* b1 = q_big + ((size_t)1 * H + h) * DCQ;
    const float* b2 = q_big + ((size_t)2 * H + h) * DCQ;
    const float* b3 = q_big + ((size_t)3 * H + h) * DCQ;
    float a0 = 0.f, a1 = 0.f, a2 = 0.f, a3 = 0.f;
    for (int q = 0; q < DCQ; ++q) {
        float w = __builtin_nontemporal_load(wp + (size_t)q * RR);
        a0 += b0[q] * w; a1 += b1[q] * w; a2 += b2[q] * w; a3 += b3[q] * w;
    }
    q_r[((size_t)0 * H + h) * RR + r] = a0;
    q_r[((size_t)1 * H + h) * RR + r] = a1;
    q_r[((size_t)2 * H + h) * RR + r] = a2;
    q_r[((size_t)3 * H + h) * RR + r] = a3;
}

// ---------------------------------------------------------------------------
// 5) qk[b,h,d] = sum_r W_kr[h,d,r] * q_r[b,h,r]   (bf16 output, WMMA A operand)
//    Folds the 137-GFLOP k_r projection into a 33-MFLOP precompute.
// ---------------------------------------------------------------------------
__global__ __launch_bounds__(256)
void qk_kernel(const float* __restrict__ q_r, const float* __restrict__ W_kr,
               __bf16* __restrict__ qk_bf) {
    __shared__ float qs[B][RR];
    const int h = blockIdx.y;
    const int d = blockIdx.x * 256 + threadIdx.x;    // 0..DC-1
    for (int i = threadIdx.x; i < B * RR; i += 256)
        qs[i >> 6][i & 63] = q_r[(size_t)(i >> 6) * (H * RR) + h * RR + (i & 63)];
    __syncthreads();
    const float* wp = W_kr + ((size_t)h * DC + d) * RR;
    float a0 = 0.f, a1 = 0.f, a2 = 0.f, a3 = 0.f;
    #pragma unroll
    for (int r = 0; r < RR; r += 4) {
        f4 w = ntload4(wp + r);
        a0 += w.x * qs[0][r] + w.y * qs[0][r + 1] + w.z * qs[0][r + 2] + w.w * qs[0][r + 3];
        a1 += w.x * qs[1][r] + w.y * qs[1][r + 1] + w.z * qs[1][r + 2] + w.w * qs[1][r + 3];
        a2 += w.x * qs[2][r] + w.y * qs[2][r + 1] + w.z * qs[2][r + 2] + w.w * qs[2][r + 3];
        a3 += w.x * qs[3][r] + w.y * qs[3][r + 1] + w.z * qs[3][r + 2] + w.w * qs[3][r + 3];
    }
    qk_bf[((size_t)0 * H + h) * DC + d] = (__bf16)a0;
    qk_bf[((size_t)1 * H + h) * DC + d] = (__bf16)a1;
    qk_bf[((size_t)2 * H + h) * DC + d] = (__bf16)a2;
    qk_bf[((size_t)3 * H + h) * DC + d] = (__bf16)a3;
}

// ---------------------------------------------------------------------------
// Fragment loaders per CDNA5 ISA 7.12.2 16-bit layouts (two 16B loads each).
// A 16x32: lanes 0-15 rows M=0..15, VGPR0-3 K=0..7 (+8 for hi lanes),
//          VGPR4-7 K=16..23 (+8 for hi lanes).
// B 32x16: lanes 0-15 col N=lane hold K=0..7,16..23; lanes 16-31 hold K=8..15,24..31.
// These operands are deliberately regular (RT) loads: kv/P/qk live in L2.
// ---------------------------------------------------------------------------
__device__ __forceinline__ v16bf load_fragA(const __bf16* rowp, int kk, int laneHi) {
    v8bf lo = *(const v8bf*)(rowp + kk + laneHi * 8);
    v8bf hi = *(const v8bf*)(rowp + kk + 16 + laneHi * 8);
    return __builtin_shufflevector(lo, hi, 0,1,2,3,4,5,6,7,8,9,10,11,12,13,14,15);
}
__device__ __forceinline__ v16bf load_fragB(const __bf16* colp, int kk, int laneHi) {
    v8bf lo = *(const v8bf*)(colp + kk + laneHi * 8);
    v8bf hi = *(const v8bf*)(colp + kk + 16 + laneHi * 8);
    return __builtin_shufflevector(lo, hi, 0,1,2,3,4,5,6,7,8,9,10,11,12,13,14,15);
}

// ---------------------------------------------------------------------------
// 6) scores S[b, h, l] = qk[b,h,:] . kv[b,l,:]   — WMMA GEMM 128x4096x512
//    grid (L/128, H/16, B), 8 waves/block, one 16x16 tile per wave.
// ---------------------------------------------------------------------------
__global__ __launch_bounds__(256)
void scores_kernel(const __bf16* __restrict__ qk_bf, const __bf16* __restrict__ kv_bf,
                   float* __restrict__ S) {
    const int wv = threadIdx.x >> 5, lane = threadIdx.x & 31;
    const int laneHi = lane >> 4, l16 = lane & 15;
    const int b = blockIdx.z, hbase = blockIdx.y * 16;
    const int Lt = blockIdx.x * 128 + wv * 16;
    const __bf16* ap = qk_bf + ((size_t)b * H + hbase + l16) * DC;       // row = head
    const __bf16* bp = kv_bf + ((size_t)b * LSEQ + Lt + l16) * DC;       // col = L pos
    v8f c = {};
    #pragma unroll 4
    for (int kk = 0; kk < DC; kk += 32) {
        v16bf a  = load_fragA(ap, kk, laneHi);
        v16bf bb = load_fragB(bp, kk, laneHi);
        c = __builtin_amdgcn_wmma_f32_16x16x32_bf16(false, a, false, bb,
                                                    (short)0, c, false, false);
    }
    float* so = S + ((size_t)(b * H + hbase + laneHi * 8)) * LSEQ + Lt + l16;
    #pragma unroll
    for (int v = 0; v < 8; ++v) so[(size_t)v * LSEQ] = c[v];
}

// ---------------------------------------------------------------------------
// 7) row softmax over L, emit bf16 probabilities (WMMA A operand for ctx)
// ---------------------------------------------------------------------------
__global__ __launch_bounds__(256)
void softmax_kernel(const float* __restrict__ S, __bf16* __restrict__ P_bf) {
    __shared__ float red[256];
    const int row = blockIdx.x;                      // b*H + h
    const float* sp = S + (size_t)row * LSEQ;
    float vals[16];
    float m = -1e30f;
    #pragma unroll
    for (int j = 0; j < 16; ++j) {
        vals[j] = sp[threadIdx.x + j * 256];
        m = fmaxf(m, vals[j]);
    }
    red[threadIdx.x] = m; __syncthreads();
    for (int off = 128; off > 0; off >>= 1) {
        if (threadIdx.x < off) red[threadIdx.x] = fmaxf(red[threadIdx.x], red[threadIdx.x + off]);
        __syncthreads();
    }
    m = red[0]; __syncthreads();
    float sum = 0.f;
    #pragma unroll
    for (int j = 0; j < 16; ++j) { vals[j] = __expf(vals[j] - m); sum += vals[j]; }
    red[threadIdx.x] = sum; __syncthreads();
    for (int off = 128; off > 0; off >>= 1) {
        if (threadIdx.x < off) red[threadIdx.x] += red[threadIdx.x + off];
        __syncthreads();
    }
    float inv = 1.0f / red[0];
    __bf16* pp = P_bf + (size_t)row * LSEQ;
    #pragma unroll
    for (int j = 0; j < 16; ++j) pp[threadIdx.x + j * 256] = (__bf16)(vals[j] * inv);
}

// ---------------------------------------------------------------------------
// 8) ctx[b,h,d] = sum_l P[b,h,l] * kv[b,l,d]   — WMMA GEMM 128x512x4096
//    grid (DC/128, H/16, B); B operand from the transposed kv copy.
// ---------------------------------------------------------------------------
__global__ __launch_bounds__(256)
void ctx_kernel(const __bf16* __restrict__ P_bf, const __bf16* __restrict__ kvT_bf,
                float* __restrict__ ctx) {
    const int wv = threadIdx.x >> 5, lane = threadIdx.x & 31;
    const int laneHi = lane >> 4, l16 = lane & 15;
    const int b = blockIdx.z, hbase = blockIdx.y * 16;
    const int dt = blockIdx.x * 128 + wv * 16;
    const __bf16* ap = P_bf + ((size_t)b * H + hbase + l16) * LSEQ;      // row = head
    const __bf16* bp = kvT_bf + ((size_t)b * DC + dt + l16) * LSEQ;      // col = d
    v8f c = {};
    #pragma unroll 4
    for (int kk = 0; kk < LSEQ; kk += 32) {
        v16bf a  = load_fragA(ap, kk, laneHi);
        v16bf bb = load_fragB(bp, kk, laneHi);
        c = __builtin_amdgcn_wmma_f32_16x16x32_bf16(false, a, false, bb,
                                                    (short)0, c, false, false);
    }
    float* co = ctx + ((size_t)(b * H + hbase + laneHi * 8)) * DC + dt + l16;
    #pragma unroll
    for (int v = 0; v < 8; ++v) co[(size_t)v * DC] = c[v];
}

// ---------------------------------------------------------------------------
// 9) ctx_lat[b,h,k] = sum_d ctx[b,h,d] * w_kc_kv[h,k,d]   (34 MB NT stream)
// ---------------------------------------------------------------------------
__global__ __launch_bounds__(128)
void ctxlat_kernel(const float* __restrict__ ctx, const float* __restrict__ w_kc_kv,
                   float* __restrict__ ctx_lat) {
    __shared__ float cs[B][DC];                      // 8 KB
    const int h = blockIdx.x;
    const int k = threadIdx.x;                       // 0..127
    for (int i = threadIdx.x; i < B * DC; i += 128)
        cs[i >> 9][i & 511] = ctx[(size_t)(i >> 9) * (H * DC) + h * DC + (i & 511)];
    __syncthreads();
    const float* wp = w_kc_kv + ((size_t)h * KDIM + k) * DC;
    float a0 = 0.f, a1 = 0.f, a2 = 0.f, a3 = 0.f;
    for (int d = 0; d < DC; d += 4) {
        f4 w = ntload4(wp + d);
        a0 += w.x * cs[0][d] + w.y * cs[0][d + 1] + w.z * cs[0][d + 2] + w.w * cs[0][d + 3];
        a1 += w.x * cs[1][d] + w.y * cs[1][d + 1] + w.z * cs[1][d + 2] + w.w * cs[1][d + 3];
        a2 += w.x * cs[2][d] + w.y * cs[2][d + 1] + w.z * cs[2][d + 2] + w.w * cs[2][d + 3];
        a3 += w.x * cs[3][d] + w.y * cs[3][d + 1] + w.z * cs[3][d + 2] + w.w * cs[3][d + 3];
    }
    ctx_lat[(size_t)0 * (H * KDIM) + h * KDIM + k] = a0;
    ctx_lat[(size_t)1 * (H * KDIM) + h * KDIM + k] = a1;
    ctx_lat[(size_t)2 * (H * KDIM) + h * KDIM + k] = a2;
    ctx_lat[(size_t)3 * (H * KDIM) + h * KDIM + k] = a3;
}

// ---------------------------------------------------------------------------
// 10) out[b,m] = sum_n ctx_lat[b,n] * Wout[m,n]   (470 MB NT stream)
// ---------------------------------------------------------------------------
__global__ __launch_bounds__(256)
void out_kernel(const float* __restrict__ ctx_lat, const float* __restrict__ Wout,
                float* __restrict__ out) {
    const int wv = threadIdx.x >> 5, lane = threadIdx.x & 31;
    const int m = blockIdx.x * 8 + wv;               // 0..DIMQ-1
    const float* wrow = Wout + (size_t)m * (H * KDIM);
    float a0 = 0.f, a1 = 0.f, a2 = 0.f, a3 = 0.f;
    for (int n = lane * 4; n < H * KDIM; n += 128) {
        f4 w  = ntload4(wrow + n);
        f4 c0 = ld4(ctx_lat + 0 * (H * KDIM) + n);
        f4 c1 = ld4(ctx_lat + 1 * (H * KDIM) + n);
        f4 c2 = ld4(ctx_lat + 2 * (H * KDIM) + n);
        f4 c3 = ld4(ctx_lat + 3 * (H * KDIM) + n);
        a0 += w.x * c0.x + w.y * c0.y + w.z * c0.z + w.w * c0.w;
        a1 += w.x * c1.x + w.y * c1.y + w.z * c1.z + w.w * c1.w;
        a2 += w.x * c2.x + w.y * c2.y + w.z * c2.z + w.w * c2.w;
        a3 += w.x * c3.x + w.y * c3.y + w.z * c3.z + w.w * c3.w;
    }
    #pragma unroll
    for (int off = 16; off > 0; off >>= 1) {
        a0 += __shfl_down(a0, off);
        a1 += __shfl_down(a1, off);
        a2 += __shfl_down(a2, off);
        a3 += __shfl_down(a3, off);
    }
    if (lane == 0) {
        out[0 * DIMQ + m] = a0;
        out[1 * DIMQ + m] = a1;
        out[2 * DIMQ + m] = a2;
        out[3 * DIMQ + m] = a3;
    }
}

// ---------------------------------------------------------------------------
extern "C" void kernel_launch(void* const* d_in, const int* in_sizes, int n_in,
                              void* d_out, int out_size, void* d_ws, size_t ws_size,
                              hipStream_t stream) {
    (void)in_sizes; (void)n_in; (void)out_size; (void)ws_size;
    const float* hidden_q = (const float*)d_in[0];
    const float* kv_c     = (const float*)d_in[1];
    const float* Wq       = (const float*)d_in[2];
    const float* w_kc_q   = (const float*)d_in[3];
    const float* w_kc_kv  = (const float*)d_in[4];
    const float* W_qr     = (const float*)d_in[5];
    const float* W_kr     = (const float*)d_in[6];
    const float* Wout     = (const float*)d_in[7];
    float* out = (float*)d_out;

    // workspace carve (all offsets 256B aligned), ~52 MB total
    char* ws = (char*)d_ws;
    size_t off = 0;
    __bf16* kv_bf   = (__bf16*)(ws + off); off += (size_t)B * LSEQ * DC * 2;   // 16 MB
    __bf16* kvT_bf  = (__bf16*)(ws + off); off += (size_t)B * DC * LSEQ * 2;   // 16 MB
    float*  q_hk    = (float*) (ws + off); off += (size_t)B * H * KDIM * 4;
    float*  q_big   = (float*) (ws + off); off += (size_t)B * H * DCQ * 4;
    float*  q_r     = (float*) (ws + off); off += (size_t)B * H * RR * 4;
    __bf16* qk_bf   = (__bf16*)(ws + off); off += (size_t)B * H * DC * 2;
    float*  S       = (float*) (ws + off); off += (size_t)B * H * LSEQ * 4;    // 8 MB
    __bf16* P_bf    = (__bf16*)(ws + off); off += (size_t)B * H * LSEQ * 2;    // 4 MB
    float*  ctx     = (float*) (ws + off); off += (size_t)B * H * DC * 4;
    float*  ctx_lat = (float*) (ws + off); off += (size_t)B * H * KDIM * 4;

    rope_kernel   <<<dim3(LSEQ, B), 512, 0, stream>>>(kv_c, kv_bf, kvT_bf);
    qhk_kernel    <<<(H * KDIM) / 8, 256, 0, stream>>>(hidden_q, Wq, q_hk);
    qbig_kernel   <<<dim3(DCQ / 256, H), 256, 0, stream>>>(q_hk, w_kc_q, q_big);
    qr_kernel     <<<H, 64, 0, stream>>>(q_big, W_qr, q_r);
    qk_kernel     <<<dim3(DC / 256, H), 256, 0, stream>>>(q_r, W_kr, qk_bf);
    scores_kernel <<<dim3(LSEQ / 128, H / 16, B), 256, 0, stream>>>(qk_bf, kv_bf, S);
    softmax_kernel<<<B * H, 256, 0, stream>>>(S, P_bf);
    ctx_kernel    <<<dim3(DC / 128, H / 16, B), 256, 0, stream>>>(P_bf, kvT_bf, ctx);
    ctxlat_kernel <<<H, 128, 0, stream>>>(ctx, w_kc_kv, ctx_lat);
    out_kernel    <<<DIMQ / 8, 256, 0, stream>>>(ctx_lat, Wout, out);
}
